// PerceiverProjector_78855599554941
// MI455X (gfx1250) — matrix operand order repaired and checked by reading, not compiled, not run on hardware
//
#include <hip/hip_runtime.h>
#include <hip/hip_bf16.h>

// ---------------------------------------------------------------------------
// PerceiverProjector on MI455X (gfx1250, wave32, WMMA).
// Pipeline: f32->bf16 convert/transposes, KV GEMM (WMMA bf16 + async-to-LDS
// staging), attention (QK^T WMMA -> softmax in LDS -> PV WMMA), output GEMM.
// ---------------------------------------------------------------------------

typedef __bf16 bf16_t;
typedef __attribute__((ext_vector_type(16))) __bf16 v16bf;
typedef __attribute__((ext_vector_type(8)))  __bf16 v8bf;
typedef __attribute__((ext_vector_type(8)))  float  v8f;

union BF16x16 { v16bf v; v8bf h[2]; };

#define B_SZ    32
#define N_SZ    1024
#define VIT_DIM 1024
#define LLM_DIM 4096
#define NUM_Q   64
#define NUM_H   16
#define HEAD_D  256
#define KV_COLS 8192
#define SCALE_F 0.0625f   // HEAD_D^-0.5

// A-matrix fragment (16x32 bf16): lane holds row (lane&15); element e in 0..15
// is K = half*16+e.
__device__ __forceinline__ v16bf load_frag_a(const bf16_t* rowbase, int half) {
    BF16x16 r;
    r.h[0] = *(const v8bf*)(rowbase + half * 16);
    r.h[1] = *(const v8bf*)(rowbase + half * 16 + 8);
    return r.v;
}

// B-matrix fragment (32x16 bf16): lane holds column (lane&15); lanes 0-15 get
// K=0..15, lanes 16-31 K=16..31 (caller folds +half*16 into `base`).
__device__ __forceinline__ v16bf load_frag_b(const bf16_t* base) {
    BF16x16 r;
    r.h[0] = *(const v8bf*)(base);
    r.h[1] = *(const v8bf*)(base + 8);
    return r.v;
}

__device__ __forceinline__ v8f wmma_bf16(v16bf a, v16bf b, v8f c) {
    return __builtin_amdgcn_wmma_f32_16x16x32_bf16(
        /*neg_a=*/false, a, /*neg_b=*/false, b,
        /*c_mod=*/(short)0, c, /*reuse_a=*/false, /*reuse_b=*/false);
}

// CDNA5 async copy: 32 bytes global -> LDS per lane (two b128 ops; the
// instruction offset applies to BOTH the LDS and global address, so a single
// address pair covers both halves). Tracked by ASYNCcnt.
__device__ __forceinline__ void async_copy_b256(void* lds_dst, const void* gsrc) {
    const unsigned ldsoff = (unsigned)(unsigned long long)lds_dst;  // addr[31:0] = LDS offset
    asm volatile(
        "global_load_async_to_lds_b128 %0, %1, off\n\t"
        "global_load_async_to_lds_b128 %0, %1, off offset:16"
        :: "v"(ldsoff), "v"(gsrc) : "memory");
}

__device__ __forceinline__ void wait_async0() {
    asm volatile("s_wait_asynccnt 0" ::: "memory");
}

// ---------------------------------------------------------------------------
// Prep kernels: f32 -> bf16 convert, and transpose+convert for weights.
// ---------------------------------------------------------------------------
__global__ __launch_bounds__(256) void cvt_bf16_kernel(
    const float* __restrict__ in, bf16_t* __restrict__ out, int n) {
    int i = blockIdx.x * 256 + threadIdx.x;
    if (i < n) out[i] = (bf16_t)in[i];
}

__global__ __launch_bounds__(256) void transpose_cvt_kernel(
    const float* __restrict__ in, bf16_t* __restrict__ out, int R, int C) {
    int i = blockIdx.x * 256 + threadIdx.x;
    if (i < R * C) {
        int r = i / C, c = i % C;
        out[(size_t)c * R + r] = (bf16_t)in[i];
    }
}

// ---------------------------------------------------------------------------
// 128x128 block-tile GEMM: C[M,N] = A[M,K] (row-major bf16) * Bt[N,K]^T.
// 8 waves/block, each owns a 32x64 sub-tile (2x4 WMMA tiles), K-step 32.
// Double-buffered LDS staging via GLOBAL_LOAD_ASYNC_TO_LDS_B128 (ASYNCcnt).
// ---------------------------------------------------------------------------
template <int KDIM, bool SPLIT_KV>
__global__ __launch_bounds__(256) void gemm128_kernel(
    const bf16_t* __restrict__ A, const bf16_t* __restrict__ Bt,
    bf16_t* __restrict__ outK, bf16_t* __restrict__ outVt,
    float* __restrict__ outF) {
    __shared__ __align__(16) bf16_t sA[2][128 * 32];
    __shared__ __align__(16) bf16_t sB[2][128 * 32];

    const int t = threadIdx.x;
    const int lane = t & 31;
    const int w = t >> 5;            // wave 0..7
    const int wm = w & 3;            // wave m-slot (4 x 32 rows)
    const int wn = w >> 2;           // wave n-slot (2 x 64 cols)
    const int half = lane >> 4;
    const int ln = lane & 15;
    const int bm = blockIdx.y * 128;
    const int bn = blockIdx.x * 128;

    // Staging: each thread async-copies 32 bytes of one 64-byte tile row.
    const int srow = t >> 1;              // 0..127
    const int soff = (t & 1) * 16;        // element offset within 32-elem row
    const bf16_t* gA = A + (size_t)(bm + srow) * KDIM + soff;
    const bf16_t* gB = Bt + (size_t)(bn + srow) * KDIM + soff;

    v8f c[2][4];
    for (int i = 0; i < 2; i++)
        for (int j = 0; j < 4; j++) c[i][j] = v8f{};

    async_copy_b256(&sA[0][srow * 32 + soff], gA);
    async_copy_b256(&sB[0][srow * 32 + soff], gB);
    wait_async0();
    __syncthreads();

    int buf = 0;
    for (int kb = 0; kb < KDIM; kb += 32) {
        const bool havenext = (kb + 32) < KDIM;
        if (havenext) {  // async prefetch into other buffer overlaps WMMA
            async_copy_b256(&sA[buf ^ 1][srow * 32 + soff], gA + kb + 32);
            async_copy_b256(&sB[buf ^ 1][srow * 32 + soff], gB + kb + 32);
        }
        v16bf af[2], bfr[4];
        for (int mt = 0; mt < 2; mt++)
            af[mt] = load_frag_a(&sA[buf][(wm * 32 + mt * 16 + ln) * 32], half);
        for (int nt = 0; nt < 4; nt++)
            bfr[nt] = load_frag_b(&sB[buf][(wn * 64 + nt * 16 + ln) * 32 + half * 16]);
        for (int mt = 0; mt < 2; mt++)
            for (int nt = 0; nt < 4; nt++)
                c[mt][nt] = wmma_bf16(af[mt], bfr[nt], c[mt][nt]);
        if (havenext) {
            wait_async0();      // own async writes to buf^1 done
            __syncthreads();    // publish to all waves
            buf ^= 1;
        }
    }

    // Store. C-frag layout: VGPR r -> row m0 + r + 8*half, col n0 + (lane&15).
    for (int mt = 0; mt < 2; mt++) {
        for (int nt = 0; nt < 4; nt++) {
            const int m0 = bm + wm * 32 + mt * 16;
            const int n0 = bn + wn * 64 + nt * 16;
            for (int r = 0; r < 8; r++) {
                const int gm = m0 + r + 8 * half;
                const int gc = n0 + ln;
                const float v = c[mt][nt][r];
                if constexpr (SPLIT_KV) {
                    if (bn < LLM_DIM) {  // K half: row-major [B*N, 4096]
                        outK[(size_t)gm * LLM_DIM + gc] = (bf16_t)v;
                    } else {             // V half: transposed [b,h,d,n]
                        const int cv = gc - LLM_DIM;
                        const int hh = cv >> 8, dd = cv & 255;
                        const int bb = gm >> 10, nn = gm & 1023;
                        outVt[((size_t)(bb * NUM_H + hh) * HEAD_D + dd) * N_SZ + nn] =
                            (bf16_t)v;
                    }
                } else {
                    outF[(size_t)gm * LLM_DIM + gc] = v;
                }
            }
        }
    }
}

// ---------------------------------------------------------------------------
// Attention: one block per (batch, head, 32-query half). 256 threads.
//  P1: scores(32x1024) = Q(32x256) @ K^T via WMMA; per k-step, all 8 K-tile
//      fragments are loaded up front (one load clause, then 16 WMMAs behind a
//      single wait). Scaled scores stored bf16 in LDS.
//  P2: two-pass softmax (f32 max/sum); probs = exp(s-max) in LDS.
//  P3: out(32x256) = P @ V via WMMA (V pre-transposed [b,h,d,n]); fragment
//      loads register-double-buffered one k-step ahead. 1/(sum+1e-9) folded
//      into the store; bf16 out row-major [b*64+q, h*256+d].
// ---------------------------------------------------------------------------
__global__ __launch_bounds__(256) void attn_kernel(
    const bf16_t* __restrict__ Qb,    // [64, 4096]
    const bf16_t* __restrict__ Kbuf,  // [B*N, 4096]
    const bf16_t* __restrict__ Vt,    // [(b*16+h)*256+d, 1024]
    bf16_t* __restrict__ AO) {        // [B*64, 4096]
    __shared__ __align__(16) bf16_t sP[32 * 1024];  // 64 KB scores/probs
    __shared__ float sPart[8 * 32];
    __shared__ float sMax[32];
    __shared__ float sInv[32];

    const int blk = blockIdx.x;
    const int qh = blk & 1;
    const int h = (blk >> 1) & 15;
    const int b = blk >> 5;
    const int q0g = qh * 32;

    const int t = threadIdx.x;
    const int lane = t & 31;
    const int w = t >> 5;
    const int half = lane >> 4;
    const int ln = lane & 15;

    // ---- Phase 1: scores = (Q K^T) * SCALE -> sP (bf16) ----
    {
        v8f acc[8][2];
        for (int nt = 0; nt < 8; nt++)
            for (int qt = 0; qt < 2; qt++) acc[nt][qt] = v8f{};
        const bf16_t* q0p = Qb + (size_t)(q0g + ln) * LLM_DIM + h * HEAD_D;
        const bf16_t* q1p = Qb + (size_t)(q0g + 16 + ln) * LLM_DIM + h * HEAD_D;
        const bf16_t* kp =
            Kbuf + (size_t)(b * N_SZ + w * 128 + ln) * LLM_DIM + h * HEAD_D + half * 16;
        for (int kb = 0; kb < HEAD_D; kb += 32) {
            // One big load clause: 2 A-frags + 8 B-frags, then 16 WMMAs.
            const v16bf a0 = load_frag_a(q0p + kb, half);
            const v16bf a1 = load_frag_a(q1p + kb, half);
            v16bf bb[8];
            for (int nt = 0; nt < 8; nt++)
                bb[nt] = load_frag_b(kp + (size_t)(nt * 16) * LLM_DIM + kb);
            for (int nt = 0; nt < 8; nt++) {
                acc[nt][0] = wmma_bf16(a0, bb[nt], acc[nt][0]);
                acc[nt][1] = wmma_bf16(a1, bb[nt], acc[nt][1]);
            }
        }
        for (int nt = 0; nt < 8; nt++) {
            const int col = w * 128 + nt * 16 + ln;
            for (int r = 0; r < 8; r++) {
                const int ql = r + 8 * half;
                sP[ql * 1024 + col] = (bf16_t)(acc[nt][0][r] * SCALE_F);
                sP[(ql + 16) * 1024 + col] = (bf16_t)(acc[nt][1][r] * SCALE_F);
            }
        }
    }
    __syncthreads();

    // ---- Phase 2: softmax over n (rows of 1024) ----
    {
        const int row = t & 31, seg = t >> 5;  // 8 segments x 128 cols
        bf16_t* pr = &sP[row * 1024 + seg * 128];
        float m = -1e30f;
        for (int cidx = 0; cidx < 128; cidx++) m = fmaxf(m, (float)pr[cidx]);
        sPart[seg * 32 + row] = m;
        __syncthreads();
        if (t < 32) {
            float mm = sPart[t];
            for (int s = 1; s < 8; s++) mm = fmaxf(mm, sPart[s * 32 + t]);
            sMax[t] = mm;
        }
        __syncthreads();
        const float rm = sMax[row];
        float sum = 0.f;
        for (int cidx = 0; cidx < 128; cidx++) {
            const float p = __expf((float)pr[cidx] - rm);
            pr[cidx] = (bf16_t)p;
            sum += p;
        }
        sPart[seg * 32 + row] = sum;
        __syncthreads();
        if (t < 32) {
            float ss = 0.f;
            for (int s = 0; s < 8; s++) ss += sPart[s * 32 + t];
            sInv[t] = 1.0f / (ss + 1e-9f);
        }
        __syncthreads();
    }

    // ---- Phase 3: out = P @ V  (each wave owns 32 of the 256 d-cols) ----
    {
        const int ds0 = w * 32;
        const size_t vbase = (size_t)(b * NUM_H + h) * HEAD_D * N_SZ;
        const bf16_t* v0p = Vt + vbase + (size_t)(ds0 + ln) * N_SZ + half * 16;
        const bf16_t* v1p = Vt + vbase + (size_t)(ds0 + 16 + ln) * N_SZ + half * 16;
        v8f c[2][2];
        for (int i = 0; i < 2; i++)
            for (int j = 0; j < 2; j++) c[i][j] = v8f{};

        // Register double-buffer: fragments for k-step kb+32 are loaded
        // before the WMMAs of kb are issued.
        v16bf a0 = load_frag_a(&sP[ln * 1024], half);
        v16bf a1 = load_frag_a(&sP[(16 + ln) * 1024], half);
        v16bf b0 = load_frag_b(v0p);
        v16bf b1 = load_frag_b(v1p);
        for (int kb = 0; kb < N_SZ; kb += 32) {
            v16bf na0, na1, nb0, nb1;
            if (kb + 32 < N_SZ) {
                na0 = load_frag_a(&sP[ln * 1024 + kb + 32], half);
                na1 = load_frag_a(&sP[(16 + ln) * 1024 + kb + 32], half);
                nb0 = load_frag_b(v0p + kb + 32);
                nb1 = load_frag_b(v1p + kb + 32);
            }
            c[0][0] = wmma_bf16(a0, b0, c[0][0]);
            c[0][1] = wmma_bf16(a0, b1, c[0][1]);
            c[1][0] = wmma_bf16(a1, b0, c[1][0]);
            c[1][1] = wmma_bf16(a1, b1, c[1][1]);
            a0 = na0; a1 = na1; b0 = nb0; b1 = nb1;
        }
        for (int qt = 0; qt < 2; qt++) {
            for (int dt = 0; dt < 2; dt++) {
                for (int r = 0; r < 8; r++) {
                    const int ql = qt * 16 + r + 8 * half;  // 0..31
                    const int d = ds0 + dt * 16 + ln;
                    const float v = c[qt][dt][r] * sInv[ql];
                    AO[((size_t)(b * NUM_Q + q0g + ql)) * LLM_DIM + h * HEAD_D + d] =
                        (bf16_t)v;
                }
            }
        }
    }
}

// ---------------------------------------------------------------------------
// Host-side launch
// ---------------------------------------------------------------------------
extern "C" void kernel_launch(void* const* d_in, const int* in_sizes, int n_in,
                              void* d_out, int out_size, void* d_ws, size_t ws_size,
                              hipStream_t stream) {
    const float* vit = (const float*)d_in[0];   // [32,1024,1024]
    const float* qry = (const float*)d_in[1];   // [1,64,4096]
    const float* wkv = (const float*)d_in[2];   // [1024,8192]
    const float* wo  = (const float*)d_in[3];   // [4096,4096]
    float* out = (float*)d_out;                 // [32*64, 4096]

    char* ws = (char*)d_ws;
    const size_t n_vit  = (size_t)B_SZ * N_SZ * VIT_DIM;
    const size_t n_q    = (size_t)NUM_Q * LLM_DIM;
    const size_t n_wkv  = (size_t)VIT_DIM * KV_COLS;
    const size_t n_wo   = (size_t)LLM_DIM * LLM_DIM;
    const size_t n_kv   = (size_t)B_SZ * N_SZ * LLM_DIM;
    const size_t n_ao   = (size_t)B_SZ * NUM_Q * LLM_DIM;

    size_t off = 0;
    bf16_t* vit_bf = (bf16_t*)(ws + off); off += n_vit * 2;
    bf16_t* q_bf   = (bf16_t*)(ws + off); off += n_q * 2;
    bf16_t* wkv_t  = (bf16_t*)(ws + off); off += n_wkv * 2;
    bf16_t* wo_t   = (bf16_t*)(ws + off); off += n_wo * 2;
    bf16_t* kbuf   = (bf16_t*)(ws + off); off += n_kv * 2;
    bf16_t* vt     = (bf16_t*)(ws + off); off += n_kv * 2;
    bf16_t* ao     = (bf16_t*)(ws + off); off += n_ao * 2;
    (void)ws_size; (void)n_in; (void)in_sizes; (void)out_size;

    cvt_bf16_kernel<<<(int)((n_vit + 255) / 256), 256, 0, stream>>>(vit, vit_bf, (int)n_vit);
    cvt_bf16_kernel<<<(int)((n_q + 255) / 256), 256, 0, stream>>>(qry, q_bf, (int)n_q);
    transpose_cvt_kernel<<<(int)((n_wkv + 255) / 256), 256, 0, stream>>>(
        wkv, wkv_t, VIT_DIM, KV_COLS);
    transpose_cvt_kernel<<<(int)((n_wo + 255) / 256), 256, 0, stream>>>(
        wo, wo_t, LLM_DIM, LLM_DIM);

    {   // KV projection: [32768,1024] @ [1024,8192] -> K (row-major) + V^T
        dim3 grid(KV_COLS / 128, (B_SZ * N_SZ) / 128);  // (64, 256)
        gemm128_kernel<VIT_DIM, true><<<grid, 256, 0, stream>>>(
            vit_bf, wkv_t, kbuf, vt, (float*)nullptr);
    }

    attn_kernel<<<B_SZ * NUM_H * 2, 256, 0, stream>>>(q_bf, kbuf, vt, ao);

    {   // output projection: [2048,4096] @ [4096,4096] -> f32 d_out
        dim3 grid(LLM_DIM / 128, (B_SZ * NUM_Q) / 128);  // (32, 16)
        gemm128_kernel<LLM_DIM, false><<<grid, 256, 0, stream>>>(
            ao, wo_t, (bf16_t*)nullptr, (bf16_t*)nullptr, out);
    }
}
